// AEGIS_72335839200051
// MI455X (gfx1250) — matrix operand order repaired and checked by reading, not compiled, not run on hardware
//
#include <hip/hip_runtime.h>
#include <math.h>

#define N_NODES 100000
#define N_EDGES 400000
#define HID 128

typedef __attribute__((ext_vector_type(16))) __bf16 v16bf;
typedef __attribute__((ext_vector_type(8)))  float  v8f;

static __device__ __forceinline__ float sigmoidf_(float x){ return 1.0f/(1.0f+__expf(-x)); }
static __device__ __forceinline__ float geluf_(float x){ return 0.5f*x*(1.0f+erff(x*0.70710678118654752f)); }

// ordered-int encoding for float atomicMax
static __device__ __forceinline__ unsigned enc_f(float f){
  unsigned u = __float_as_uint(f);
  return (u & 0x80000000u) ? ~u : (u | 0x80000000u);
}
static __device__ __forceinline__ float dec_f(unsigned k){
  unsigned u = (k & 0x80000000u) ? (k & 0x7fffffffu) : ~k;
  return __uint_as_float(u);
}

// ---------------- utility kernels ----------------
__global__ void k_fill_f32(float* p, float v, int n){ int i=blockIdx.x*256+threadIdx.x; if(i<n) p[i]=v; }
__global__ void k_fill_u32(unsigned* p, unsigned v, int n){ int i=blockIdx.x*256+threadIdx.x; if(i<n) p[i]=v; }
__global__ void k_copy(const float* __restrict__ a, float* __restrict__ b, int n){
  int i=blockIdx.x*256+threadIdx.x; if(i<n) b[i]=a[i];
}

// ---------------- weight pre-swizzle: f32 [K x 128] -> bf16 WMMA B-fragment order ----
// flat idx = ((kt*8 + t)*32 + lane)*16 + j ; value = W[(kt*32 + (j>>3)*16 + (lane>>4)*8 + (j&7))*128 + t*16 + (lane&15)]
__global__ void k_wswizzle(const float* __restrict__ W, __bf16* __restrict__ Wf, int K){
  int idx = blockIdx.x*256+threadIdx.x;
  if (idx >= K*HID) return;
  int j  = idx & 15;
  int ln = (idx >> 4) & 31;
  int t  = (idx >> 9) & 7;
  int kt = idx >> 12;
  int k  = kt*32 + (j>>3)*16 + (ln>>4)*8 + (j&7);
  int n  = t*16 + (ln&15);
  Wf[idx] = (__bf16)W[(size_t)k*HID + n];
}

// ---------------- WMMA GEMM helpers ----------------
static __device__ __forceinline__ v16bf load_a_f4(const float* __restrict__ arow, int k0, int hi){
  const float4* p0 = (const float4*)(arow + k0 + hi*8);        // j = 0..7
  const float4* p1 = (const float4*)(arow + k0 + 16 + hi*8);   // j = 8..15
  float4 a0 = p0[0], a1 = p0[1], a2 = p1[0], a3 = p1[1];
  v16bf a;
  a[0]=(__bf16)a0.x; a[1]=(__bf16)a0.y; a[2]=(__bf16)a0.z; a[3]=(__bf16)a0.w;
  a[4]=(__bf16)a1.x; a[5]=(__bf16)a1.y; a[6]=(__bf16)a1.z; a[7]=(__bf16)a1.w;
  a[8]=(__bf16)a2.x; a[9]=(__bf16)a2.y; a[10]=(__bf16)a2.z; a[11]=(__bf16)a2.w;
  a[12]=(__bf16)a3.x; a[13]=(__bf16)a3.y; a[14]=(__bf16)a3.z; a[15]=(__bf16)a3.w;
  return a;
}

// epilogue: optional bias (per-col), residual (per-elem), row LayerNorm, GELU, store
static __device__ __forceinline__ void tile_epilogue(
    v8f (&acc)[8], const float* __restrict__ bias, const float* __restrict__ resid,
    const float* __restrict__ ln_g, const float* __restrict__ ln_b, int act,
    float* __restrict__ out, int m0, int l, int hi)
{
  if (bias) {
#pragma unroll
    for (int t=0;t<8;++t)
#pragma unroll
      for (int r=0;r<8;++r) acc[t][r] += bias[t*16+l];
  }
  if (resid) {
#pragma unroll
    for (int t=0;t<8;++t)
#pragma unroll
      for (int r=0;r<8;++r) acc[t][r] += resid[(size_t)(m0 + (hi? r+8:r))*HID + t*16 + l];
  }
  if (ln_g) {
#pragma unroll
    for (int r=0;r<8;++r) {
      float s=0.f;
#pragma unroll
      for (int t=0;t<8;++t) s += acc[t][r];
#pragma unroll
      for (int m=1;m<16;m<<=1) s += __shfl_xor(s, m, 16);
      float mean = s*(1.0f/128.0f);
      float ss=0.f;
#pragma unroll
      for (int t=0;t<8;++t){ float q=acc[t][r]-mean; ss+=q*q; }
#pragma unroll
      for (int m=1;m<16;m<<=1) ss += __shfl_xor(ss, m, 16);
      float rstd = rsqrtf(ss*(1.0f/128.0f)+1e-5f);
#pragma unroll
      for (int t=0;t<8;++t) acc[t][r] = (acc[t][r]-mean)*rstd*ln_g[t*16+l] + ln_b[t*16+l];
    }
  }
  if (act==1) {
#pragma unroll
    for (int t=0;t<8;++t)
#pragma unroll
      for (int r=0;r<8;++r) acc[t][r] = geluf_(acc[t][r]);
  }
#pragma unroll
  for (int t=0;t<8;++t)
#pragma unroll
    for (int r=0;r<8;++r)
      out[(size_t)(m0 + (hi? r+8:r))*HID + t*16 + l] = acc[t][r];
}

// C[M x 128] = A[M x K] @ W[K x 128]; Wf is bf16 fragment-ordered, staged in LDS per block.
__global__ void k_gemm_n128(const float* __restrict__ A, const __bf16* __restrict__ Wf,
                            const float* __restrict__ bias, const float* __restrict__ resid,
                            float* __restrict__ out, int M, int K,
                            const float* __restrict__ ln_g, const float* __restrict__ ln_b, int act)
{
  extern __shared__ __align__(16) __bf16 ldsW[];
  {
    const uint4* s4 = (const uint4*)Wf;
    uint4* d4 = (uint4*)ldsW;
    int n16 = K*HID/8;                       // (K*128*2)/16 bytes
    for (int i = threadIdx.x; i < n16; i += 256) d4[i] = s4[i];
  }
  __syncthreads();
  int wave = threadIdx.x >> 5, lane = threadIdx.x & 31;
  int m0 = (blockIdx.x*8 + wave)*16;
  if (m0 >= M) return;
  int l = lane & 15, hi = lane >> 4;
  v8f acc[8] = {};
  const float* arow = A + (size_t)(m0 + l)*K;
  for (int kt = 0; kt < (K>>5); ++kt) {
    v16bf a = load_a_f4(arow, kt*32, hi);
    const __bf16* wb = ldsW + ((size_t)(kt*8)*32 + lane)*16;
#pragma unroll
    for (int t=0;t<8;++t) {
      v16bf b = *((const v16bf*)(wb + (size_t)t*32*16));
      acc[t] = __builtin_amdgcn_wmma_f32_16x16x32_bf16(false, a, false, b, (short)0, acc[t], false, false);
    }
  }
  tile_epilogue(acc, bias, resid, ln_g, ln_b, act, out, m0, l, hi);
}

// Fused edge GEMM: A row e = [x_mi[src[e]], x_mi[dst[e]], e_mi[e]] (K=384), Wf bf16 in LDS.
__global__ void k_gemm_edge3(const float* __restrict__ xm, const float* __restrict__ em,
                             const int* __restrict__ src, const int* __restrict__ dst,
                             const __bf16* __restrict__ Wf, const float* __restrict__ bias,
                             float* __restrict__ out, int E,
                             const float* __restrict__ ln_g, const float* __restrict__ ln_b, int act)
{
  extern __shared__ __align__(16) __bf16 ldsW[];
  {
    const uint4* s4 = (const uint4*)Wf;
    uint4* d4 = (uint4*)ldsW;
    int n16 = 384*HID/8;
    for (int i = threadIdx.x; i < n16; i += 256) d4[i] = s4[i];
  }
  __syncthreads();
  int wave = threadIdx.x >> 5, lane = threadIdx.x & 31;
  int m0 = (blockIdx.x*8 + wave)*16;
  if (m0 >= E) return;
  int l = lane & 15, hi = lane >> 4;
  int e = m0 + l;
  const float* part[3];
  part[0] = xm + (size_t)src[e]*HID;
  part[1] = xm + (size_t)dst[e]*HID;
  part[2] = em + (size_t)e*HID;
  v8f acc[8] = {};
#pragma unroll
  for (int p=0;p<3;++p) {
    const float* arow = part[p];
#pragma unroll
    for (int kk=0; kk<4; ++kk) {
      int kt = p*4 + kk;
      v16bf a = load_a_f4(arow, kk*32, hi);
      const __bf16* wb = ldsW + ((size_t)(kt*8)*32 + lane)*16;
#pragma unroll
      for (int t=0;t<8;++t) {
        v16bf b = *((const v16bf*)(wb + (size_t)t*32*16));
        acc[t] = __builtin_amdgcn_wmma_f32_16x16x32_bf16(false, a, false, b, (short)0, acc[t], false, false);
      }
    }
  }
  tile_epilogue(acc, bias, nullptr, ln_g, ln_b, act, out, m0, l, hi);
}

// ---------------- graph structure ----------------
__global__ void k_degrees(const int* __restrict__ src, const int* __restrict__ dst,
                          float* outdeg, float* indeg, int E){
  int i = blockIdx.x*256+threadIdx.x;
  if (i < E){ atomicAdd(&outdeg[src[i]], 1.0f); atomicAdd(&indeg[dst[i]], 1.0f); }
}
__global__ void k_entropy(const int* __restrict__ src, const float* __restrict__ outdeg,
                          float* ent, int E){
  __shared__ float sh[256];
  int i = blockIdx.x*256+threadIdx.x;
  float v = 0.f;
  if (i < E){ float p = 1.0f/(outdeg[src[i]]+1e-6f); v = -p*__logf(p+1e-6f); }
  sh[threadIdx.x]=v; __syncthreads();
  for (int s=128;s>0;s>>=1){ if (threadIdx.x<s) sh[threadIdx.x]+=sh[threadIdx.x+s]; __syncthreads(); }
  if (threadIdx.x==0) atomicAdd(ent, sh[0]);
}

// ---------------- macro branch (SAGE mean aggregate) ----------------
__global__ void k_scatter_sum(const float* __restrict__ xin, const int* __restrict__ src,
                              const int* __restrict__ dst, float* agg, int E){
  int gid = blockIdx.x*256+threadIdx.x;
  if (gid >= E*HID) return;
  int e = gid >> 7, c = gid & 127;
  atomicAdd(&agg[(size_t)dst[e]*HID + c], xin[(size_t)src[e]*HID + c]);
}
__global__ void k_scale_rows(float* x, const float* __restrict__ indeg, int n){
  int gid = blockIdx.x*256+threadIdx.x;
  if (gid < n) x[gid] /= fmaxf(indeg[gid>>7], 1.0f);
}

// ---------------- GraphNorm (per-feature stats over nodes) ----------------
__global__ void k_col_accum(const float* __restrict__ x, float* colsum, int M){
  int c = threadIdx.x;                     // blockDim = 128
  int r0 = blockIdx.x*64, r1 = min(r0+64, M);
  float s = 0.f;
  for (int r=r0; r<r1; ++r) s += x[(size_t)r*HID + c];
  atomicAdd(&colsum[c], s);
}
__global__ void k_col_sq(const float* __restrict__ x, const float* __restrict__ colsum,
                         const float* __restrict__ alpha, float* colvar, int M){
  int c = threadIdx.x;
  int r0 = blockIdx.x*64, r1 = min(r0+64, M);
  float m = colsum[c]/(float)M, a = alpha[c];
  float s = 0.f;
  for (int r=r0; r<r1; ++r){ float q = x[(size_t)r*HID+c] - a*m; s += q*q; }
  atomicAdd(&colvar[c], s);
}
__global__ void k_gn_apply(const float* __restrict__ x, const float* __restrict__ colsum,
                           const float* __restrict__ colvar, const float* __restrict__ w,
                           const float* __restrict__ b, const float* __restrict__ alpha,
                           const float* __restrict__ resid_after, float* out, int M){
  int gid = blockIdx.x*256+threadIdx.x;
  if (gid >= M*HID) return;
  int c = gid & 127;
  float m = colsum[c]/(float)M, var = colvar[c]/(float)M;
  float q = x[gid] - alpha[c]*m;
  float y = geluf_(w[c]*q*rsqrtf(var+1e-5f) + b[c]);
  if (resid_after) y += resid_after[gid];
  out[gid] = y;
}

// ---------------- micro attention ----------------
__global__ void k_attn_score(const float* __restrict__ q, const float* __restrict__ k,
                             const float* __restrict__ ep, const int* __restrict__ src,
                             const int* __restrict__ dst, float* score, unsigned* smax, int E){
  int gid = blockIdx.x*256+threadIdx.x;
  if (gid >= E*8) return;
  int e = gid >> 3, h = gid & 7;
  int s = src[e], d = dst[e];
  const float* qp  = q  + (size_t)d*HID + h*16;
  const float* kp  = k  + (size_t)s*HID + h*16;
  const float* epp = ep + (size_t)e*HID + h*16;
  float acc = 0.f;
#pragma unroll
  for (int i=0;i<16;++i) acc += qp[i]*(kp[i]+epp[i]);
  acc *= 0.25f;                            // 1/sqrt(16)
  score[gid] = acc;
  atomicMax(&smax[(size_t)d*8 + h], enc_f(acc));
}
__global__ void k_attn_exp(const int* __restrict__ dst, float* score,
                           const unsigned* __restrict__ smax, float* den, int E){
  int gid = blockIdx.x*256+threadIdx.x;
  if (gid >= E*8) return;
  int e = gid >> 3, h = gid & 7, d = dst[e];
  float ex = __expf(score[gid] - dec_f(smax[(size_t)d*8+h]));
  score[gid] = ex;
  atomicAdd(&den[(size_t)d*8+h], ex);
}
__global__ void k_attn_msg(const float* __restrict__ ex, const float* __restrict__ den,
                           const float* __restrict__ v, const float* __restrict__ ep,
                           const int* __restrict__ src, const int* __restrict__ dst,
                           float* outb, int E){
  int gid = blockIdx.x*256+threadIdx.x;
  if (gid >= E*HID) return;
  int e = gid >> 7, c = gid & 127, h = c >> 4;
  int s = src[e], d = dst[e];
  float alpha = ex[(size_t)e*8+h] / (den[(size_t)d*8+h] + 1e-16f);
  atomicAdd(&outb[(size_t)d*HID + c], alpha*(v[(size_t)s*HID + c] + ep[gid]));
}

// e_mi = LN(upd * sigmoid(gate) + e_mi); one wave per edge row
__global__ void k_edge_combine(const float* __restrict__ upd, const float* __restrict__ gatep,
                               float* em, const float* __restrict__ g, const float* __restrict__ b,
                               int E){
  int row = blockIdx.x*8 + (threadIdx.x>>5);
  if (row >= E) return;
  int lane = threadIdx.x & 31;
  float v[4]; float s = 0.f;
#pragma unroll
  for (int j=0;j<4;++j){
    size_t idx = (size_t)row*HID + lane + j*32;
    float t = upd[idx]*sigmoidf_(gatep[idx]) + em[idx];
    v[j]=t; s+=t;
  }
#pragma unroll
  for (int m=1;m<32;m<<=1) s += __shfl_xor(s, m, 32);
  float mean = s*(1.0f/128.0f);
  float ss = 0.f;
#pragma unroll
  for (int j=0;j<4;++j){ float q=v[j]-mean; ss+=q*q; }
#pragma unroll
  for (int m=1;m<32;m<<=1) ss += __shfl_xor(ss, m, 32);
  float rstd = rsqrtf(ss*(1.0f/128.0f)+1e-5f);
#pragma unroll
  for (int j=0;j<4;++j){
    int c = lane + j*32;
    em[(size_t)row*HID + c] = (v[j]-mean)*rstd*g[c] + b[c];
  }
}

// ---------------- final gating ----------------
__global__ void k_gate_h1(const float* __restrict__ xma, const float* __restrict__ xmi,
                          const float* __restrict__ ent, const float* __restrict__ w1,
                          const float* __restrict__ b1, float* h1, int N){
  int gid = blockIdx.x*256+threadIdx.x;
  if (gid >= N*64) return;
  int n = gid >> 6, j = gid & 63;
  float gent = ent[0]*(1.0f/(float)N_EDGES);
  float s = b1[j] + gent*w1[256*64 + j];
  const float* xa = xma + (size_t)n*HID;
  const float* xb = xmi + (size_t)n*HID;
  for (int c=0;c<HID;++c) s += xa[c]*w1[c*64+j] + xb[c]*w1[(HID+c)*64+j];
  h1[gid] = s*sigmoidf_(s);                // silu
}
__global__ void k_gate_a(const float* __restrict__ h1, const float* __restrict__ w2,
                         const float* __restrict__ b2, float* avec, int N){
  int n = blockIdx.x*256+threadIdx.x;
  if (n >= N) return;
  float s = b2[0];
  const float* hp = h1 + (size_t)n*64;
#pragma unroll
  for (int j=0;j<64;++j) s += hp[j]*w2[j];
  avec[n] = sigmoidf_(s);
}
__global__ void k_final_mix(const float* __restrict__ xma, const float* __restrict__ xmi,
                            const float* __restrict__ avec, float* out, int N){
  int gid = blockIdx.x*256+threadIdx.x;
  if (gid >= N*HID) return;
  float a = avec[gid>>7];
  out[gid] = a*xma[gid] + (1.0f-a)*xmi[gid];
}

// ---------------- launcher ----------------
extern "C" void kernel_launch(void* const* d_in, const int* in_sizes, int n_in,
                              void* d_out, int out_size, void* d_ws, size_t ws_size,
                              hipStream_t stream)
{
  (void)in_sizes; (void)n_in; (void)out_size; (void)ws_size;
  const float* x     = (const float*)d_in[0];
  const float* eattr = (const float*)d_in[1];
  const int*   eidx  = (const int*)d_in[2];
  const int*   src   = eidx;
  const int*   dst   = eidx + N_EDGES;
  auto F = [&](int i){ return (const float*)d_in[i]; };

  // workspace carve-up (floats; every block but ent (1 float) is a multiple of 4 floats)
  float* wsf = (float*)d_ws;
  size_t o = 0;
  auto alloc = [&](size_t n){ float* p = wsf + o; o += n; return p; };
  const size_t NB = (size_t)N_NODES*HID, EB = (size_t)N_EDGES*HID;
  float* xbase  = alloc(NB); float* xmacro = alloc(NB); float* xmicro = alloc(NB);
  float* nq     = alloc(NB); float* nk     = alloc(NB); float* nv     = alloc(NB);
  float* ntmp   = alloc(NB); float* ntmp2  = alloc(NB);
  float* emicro = alloc(EB); float* eproj  = alloc(EB);
  float* ebuf1  = alloc(EB); float* ebuf2  = alloc(EB);
  float* score  = alloc((size_t)N_EDGES*8);
  unsigned* smax = (unsigned*)alloc((size_t)N_NODES*8);
  float* den    = alloc((size_t)N_NODES*8);
  float* outdeg = alloc(N_NODES); float* indeg = alloc(N_NODES);
  float* colsum = alloc(HID);     float* colvar = alloc(HID);
  float* h1     = alloc((size_t)N_NODES*64);
  float* avec   = alloc(N_NODES); float* ent = alloc(1);

  // bf16 fragment-ordered weight buffers (elems/2 floats each)
  auto allocbf = [&](size_t elems){ return (__bf16*)alloc(elems/2); };
  __bf16* wf_ne   = allocbf(128*HID);   // node_enc.w
  __bf16* wf_ee   = allocbf(64*HID);    // edge_enc.w
  __bf16* wf_mac[2][2];                 // lin_l, lin_r
  for (int li=0; li<2; ++li){ wf_mac[li][0]=allocbf(128*HID); wf_mac[li][1]=allocbf(128*HID); }
  __bf16* wf_mic[2][8];                 // wq,wk,wv,we,op_w,mlp_w2 (K=128), mlp_w1,gate_w (K=384)
  for (int li=0; li<2; ++li){
    for (int j=0;j<6;++j) wf_mic[li][j]=allocbf(128*HID);
    wf_mic[li][6]=allocbf(384*HID); wf_mic[li][7]=allocbf(384*HID);
  }

  auto fill  = [&](float* p, float v, size_t n){
    k_fill_f32<<<(int)((n+255)/256),256,0,stream>>>(p,v,(int)n); };
  auto fillu = [&](unsigned* p, unsigned v, size_t n){
    k_fill_u32<<<(int)((n+255)/256),256,0,stream>>>(p,v,(int)n); };
  auto copy  = [&](const float* a, float* b, size_t n){
    k_copy<<<(int)((n+255)/256),256,0,stream>>>(a,b,(int)n); };
  auto wswz  = [&](const float* W, __bf16* Wf, int K){
    k_wswizzle<<<(K*HID+255)/256,256,0,stream>>>(W,Wf,K); };
  auto gemm  = [&](const float* A, const __bf16* Wf, const float* bias, const float* resid,
                   float* out, int M, int K, const float* g, const float* b, int act){
    k_gemm_n128<<<dim3((M+127)/128),dim3(256),(size_t)K*HID*2,stream>>>(A,Wf,bias,resid,out,M,K,g,b,act); };
  auto graphnorm = [&](const float* xin, const float* gw, const float* gb, const float* ga,
                       const float* resid_after, float* outp){
    fill(colsum,0.f,HID);
    k_col_accum<<<(N_NODES+63)/64,128,0,stream>>>(xin,colsum,N_NODES);
    fill(colvar,0.f,HID);
    k_col_sq<<<(N_NODES+63)/64,128,0,stream>>>(xin,colsum,ga,colvar,N_NODES);
    k_gn_apply<<<(int)((NB+255)/256),256,0,stream>>>(xin,colsum,colvar,gw,gb,ga,resid_after,outp,N_NODES);
  };

  // ---- pre-swizzle all WMMA weights to bf16 fragment order ----
  wswz(F(3), wf_ne, 128);
  wswz(F(7), wf_ee, 64);
  for (int li=0; li<2; ++li){
    int mb = 11 + 6*li;
    wswz(F(mb+0), wf_mac[li][0], 128);
    wswz(F(mb+2), wf_mac[li][1], 128);
  }
  for (int li=0; li<2; ++li){
    int mb = 23 + 19*li;
    wswz(F(mb+0), wf_mic[li][0], 128);   // wq
    wswz(F(mb+1), wf_mic[li][1], 128);   // wk
    wswz(F(mb+2), wf_mic[li][2], 128);   // wv
    wswz(F(mb+3), wf_mic[li][3], 128);   // we
    wswz(F(mb+4), wf_mic[li][4], 128);   // op_w
    wswz(F(mb+13),wf_mic[li][5], 128);   // mlp_w2
    wswz(F(mb+9), wf_mic[li][6], 384);   // mlp_w1
    wswz(F(mb+15),wf_mic[li][7], 384);   // gate_w
  }

  // ---- structural entropy + degrees ----
  fill(outdeg,0.f,N_NODES); fill(indeg,0.f,N_NODES); fill(ent,0.f,1);
  k_degrees<<<(N_EDGES+255)/256,256,0,stream>>>(src,dst,outdeg,indeg,N_EDGES);
  k_entropy<<<(N_EDGES+255)/256,256,0,stream>>>(src,outdeg,ent,N_EDGES);

  // ---- encoders (GEMM + bias + rowwise LN fused) ----
  gemm(x,     wf_ne, F(4), nullptr, xbase,  N_NODES, 128, F(5), F(6), 0);
  gemm(eattr, wf_ee, F(8), nullptr, emicro, N_EDGES,  64, F(9), F(10),0);
  copy(xbase, xmacro, NB);
  copy(xbase, xmicro, NB);

  // ---- macro: 2x SAGEConv(mean) + GraphNorm + GELU + residual ----
  for (int li=0; li<2; ++li){
    int mb = 11 + 6*li;
    fill(ntmp,0.f,NB);
    k_scatter_sum<<<(int)((EB+255)/256),256,0,stream>>>(xmacro,src,dst,ntmp,N_EDGES);
    k_scale_rows<<<(int)((NB+255)/256),256,0,stream>>>(ntmp,indeg,(int)NB);
    gemm(ntmp,   wf_mac[li][0], F(mb+1), nullptr, ntmp2, N_NODES, 128, nullptr,nullptr,0);
    gemm(xmacro, wf_mac[li][1], nullptr, ntmp2,   ntmp2, N_NODES, 128, nullptr,nullptr,0);
    graphnorm(ntmp2, F(mb+3), F(mb+4), F(mb+5), /*resid_after=*/xmacro, xmacro);
  }

  // ---- micro: 2x edge-augmented attention + edge updater ----
  for (int li=0; li<2; ++li){
    int mb = 23 + 19*li;
    gemm(xmicro, wf_mic[li][0], nullptr, nullptr, nq,    N_NODES, 128, nullptr,nullptr,0);
    gemm(xmicro, wf_mic[li][1], nullptr, nullptr, nk,    N_NODES, 128, nullptr,nullptr,0);
    gemm(xmicro, wf_mic[li][2], nullptr, nullptr, nv,    N_NODES, 128, nullptr,nullptr,0);
    gemm(emicro, wf_mic[li][3], nullptr, nullptr, eproj, N_EDGES, 128, nullptr,nullptr,0);
    fillu(smax,0u,(size_t)N_NODES*8);
    fill(den,0.f,(size_t)N_NODES*8);
    k_attn_score<<<(N_EDGES*8+255)/256,256,0,stream>>>(nq,nk,eproj,src,dst,score,smax,N_EDGES);
    k_attn_exp  <<<(N_EDGES*8+255)/256,256,0,stream>>>(dst,score,smax,den,N_EDGES);
    fill(ntmp,0.f,NB);
    k_attn_msg  <<<(int)((EB+255)/256),256,0,stream>>>(score,den,nv,eproj,src,dst,ntmp,N_EDGES);
    // out@op_w + op_b + residual(x_micro), then GraphNorm + GELU
    gemm(ntmp, wf_mic[li][4], F(mb+5), xmicro, ntmp2, N_NODES, 128, nullptr,nullptr,0);
    graphnorm(ntmp2, F(mb+6), F(mb+7), F(mb+8), /*resid_after=*/nullptr, xmicro);
    // edge updater: gathered cat-GEMMs (K=384), LN+GELU fused for mlp path
    k_gemm_edge3<<<dim3((N_EDGES+127)/128),dim3(256),(size_t)384*HID*2,stream>>>(
        xmicro, emicro, src, dst, wf_mic[li][6], F(mb+10), ebuf1, N_EDGES, F(mb+11), F(mb+12), 1);
    k_gemm_edge3<<<dim3((N_EDGES+127)/128),dim3(256),(size_t)384*HID*2,stream>>>(
        xmicro, emicro, src, dst, wf_mic[li][7], F(mb+16), ebuf2, N_EDGES, nullptr, nullptr, 0);
    gemm(ebuf1, wf_mic[li][5], F(mb+14), nullptr, eproj, N_EDGES, 128, nullptr,nullptr,0);
    k_edge_combine<<<(N_EDGES+7)/8,256,0,stream>>>(eproj, ebuf2, emicro, F(mb+17), F(mb+18), N_EDGES);
  }

  // ---- spatial entropy gating ----
  k_gate_h1 <<<(N_NODES*64+255)/256,256,0,stream>>>(xmacro,xmicro,ent,F(61),F(62),h1,N_NODES);
  k_gate_a  <<<(N_NODES+255)/256,256,0,stream>>>(h1,F(63),F(64),avec,N_NODES);
  k_final_mix<<<(int)((NB+255)/256),256,0,stream>>>(xmacro,xmicro,avec,(float*)d_out,N_NODES);
}
